// VectorQuantizer_2388001817054
// MI455X (gfx1250) — matrix operand order repaired and checked by reading, not compiled, not run on hardware
//
#include <hip/hip_runtime.h>
#include <hip/hip_bf16.h>
#include <math.h>

typedef __attribute__((ext_vector_type(16))) _Float16     v16h;
typedef __attribute__((ext_vector_type(8)))  float        v8f;
typedef __attribute__((ext_vector_type(4)))  unsigned int u32x4;
typedef __attribute__((ext_vector_type(8)))  int          i32x8;
typedef __attribute__((ext_vector_type(4)))  int          i32x4;

#define VQ_N 32768
#define VQ_D 256
#define VQ_K 4096
#define ROWS_PER_WG 128
#define WAVES 8
#define NT (VQ_K / 16)            // 256 code tiles
#define ND (VQ_N * VQ_D)          // quantized_st elements; scalars at ND..ND+2; indices at ND+3..

// ---------------------------------------------------------------------------
// Kernel 1: convert codebook f32 -> f16 (2MB in ws; stays L2-resident)
// ---------------------------------------------------------------------------
__global__ __launch_bounds__(256) void k_cvt_emb(const float* __restrict__ emb,
                                                 _Float16* __restrict__ emb_h) {
    int i = blockIdx.x * 256 + threadIdx.x;      // grid covers K*D
    emb_h[i] = (_Float16)emb[i];
}

// ---------------------------------------------------------------------------
// Kernel 2: per-code ||e||^2 bias, and zero the histogram
// ---------------------------------------------------------------------------
__global__ __launch_bounds__(256) void k_esq(const float* __restrict__ emb,
                                             float* __restrict__ esq,
                                             int* __restrict__ hist) {
    int k = blockIdx.x * 256 + threadIdx.x;      // grid covers K
    const float* e = emb + (size_t)k * VQ_D;
    float s = 0.0f;
    #pragma unroll 8
    for (int d = 0; d < VQ_D; ++d) { float v = e[d]; s += v * v; }
    esq[k] = s;
    hist[k] = 0;
}

// ---------------------------------------------------------------------------
// TDM: DMA one 16x256 f16 emb tile from global into LDS (D# per ISA 08 §8).
// 2-D tensor == the tile itself: dim0=256 elems of 2B, dim1=16 rows, stride=256.
// Groups 2/3 zero (no dims 2..4). type=2, count=1.
// 6-arg builtin form (clang-23 / therock-10.0 headers).
// ---------------------------------------------------------------------------
__device__ __forceinline__ void tdm_load_tile(const _Float16* gsrc, unsigned lds_off) {
    unsigned long long ga = (unsigned long long)(size_t)gsrc;
    u32x4 g0;
    g0[0] = 1u;                                            // count=1, user D#
    g0[1] = lds_off;                                       // lds_addr (bytes)
    g0[2] = (unsigned)(ga & 0xFFFFFFFFu);                  // global_addr[31:0]
    g0[3] = (unsigned)((ga >> 32) & 0x01FFFFFFu) | (2u << 30); // addr[56:32] | type=2
    i32x8 g1;
    g1[0] = (1 << 16);        // workgroup_mask=0, data_size=1 (2 bytes)
    g1[1] = (int)(256u << 16);// tensor_dim0 = 256 (bits 79:48, low half here)
    g1[2] = (int)(16u << 16); // tensor_dim0 hi=0 | tensor_dim1 = 16
    g1[3] = (int)(256u << 16);// tensor_dim1 hi=0 | tile_dim0 = 256
    g1[4] = 16;               // tile_dim1 = 16, tile_dim2 = 0
    g1[5] = 256;              // tensor_dim0_stride = 256 (low 32)
    g1[6] = 0;                // stride0 hi | stride1 lo
    g1[7] = 0;
    i32x4 z4 = {0, 0, 0, 0};
    i32x8 z8 = {0, 0, 0, 0, 0, 0, 0, 0};
    __builtin_amdgcn_tensor_load_to_lds(g0, g1, z4, z4, z8, 0);
}

// ---------------------------------------------------------------------------
// Kernel 3: fused score-GEMM (WMMA f16->f32) + running argmin + gather +
//           loss partial + histogram.
// Score(n,k) = ||e_k||^2 - 2 * x_n . e_k   (||x||^2 omitted: row-constant)
// A fragment holds (-2*x) in f16 so WMMA computes the score directly.
// Emb tiles arrive via TDM double-buffering (TENSORcnt).
// ---------------------------------------------------------------------------
__global__ __launch_bounds__(256) void vq_main(const float*    __restrict__ x,
                                               const float*    __restrict__ emb,
                                               const _Float16* __restrict__ emb_h,
                                               const float*    __restrict__ esq,
                                               const float*    __restrict__ mask,
                                               int*            __restrict__ hist,
                                               float*          __restrict__ partial,
                                               float*          __restrict__ out) {
    __shared__ __align__(32) _Float16 sB[2][16 * VQ_D];  // 2 x 8KB ping-pong emb tiles
    __shared__ float sEsq[VQ_K];                         // 16KB ||e||^2 table
    __shared__ int   sRowIdx[ROWS_PER_WG];
    __shared__ float sRed[WAVES];

    const int tid  = threadIdx.x;
    const int w    = tid >> 5;          // wave id 0..7
    const int lane = tid & 31;
    const int half = lane >> 4;         // 0: lanes 0-15, 1: lanes 16-31
    const int col  = lane & 15;         // code column within tile / A-row id
    const int rowBase = blockIdx.x * ROWS_PER_WG + w * 16;

    const unsigned sb_off0 = (unsigned)(size_t)(&sB[0][0]);
    const unsigned sb_off1 = (unsigned)(size_t)(&sB[1][0]);

    // Kick off TDM for tile 0 immediately; overlap with A/esq staging below.
    if (w == 0) tdm_load_tile(emb_h, sb_off0);

    // Stage ||e||^2 into LDS (vectorized, 4 iterations of 256 threads)
    for (int i = tid; i < VQ_K / 4; i += 256)
        ((float4*)sEsq)[i] = ((const float4*)esq)[i];

    // ---- Preload A fragments: 16 rows x 256 (f16, scaled by -2) -> 8 x v16h
    // Layout (16-bit A 16x32, contiguous K to match B): lane<16 holds M=lane, K = d0+{0..15};
    // lane>=16 holds M=lane-16, K = d0+{16..31}.
    v16h a[8];
    {
        const float* xrow = x + (size_t)(rowBase + col) * VQ_D;
        #pragma unroll
        for (int c = 0; c < 8; ++c) {
            const int d0 = c * 32;
            const int o0 = d0 + half * 16;
            const int o1 = d0 + half * 16 + 8;
            #pragma unroll
            for (int j = 0; j < 8; ++j) a[c][j]     = (_Float16)(-2.0f * xrow[o0 + j]);
            #pragma unroll
            for (int j = 0; j < 8; ++j) a[c][8 + j] = (_Float16)(-2.0f * xrow[o1 + j]);
        }
    }

    // Running per-lane argmin. D-matrix layout: lane<16 owns rows M=0..7 (reg v -> M=v),
    // lane>=16 owns rows M=8..15 (reg v -> M=8+v); column N = lane&15.
    float minv[8];
    int   mini[8];
    #pragma unroll
    for (int v = 0; v < 8; ++v) { minv[v] = 3.402823466e38f; mini[v] = 0; }

    if (w == 0) __builtin_amdgcn_s_wait_tensorcnt(0);   // tile 0 resident
    __syncthreads();

    // ---- Main loop over 256 code tiles of 16 codes (TDM double-buffered)
    for (int kt = 0; kt < NT; ++kt) {
        const int k0 = kt << 4;
        // DMA next tile into the other buffer while we compute on this one
        if (w == 0 && kt + 1 < NT)
            tdm_load_tile(emb_h + (size_t)(k0 + 16) * VQ_D,
                          ((kt + 1) & 1) ? sb_off1 : sb_off0);

        const _Float16* curB = &sB[kt & 1][0];
        const float bias = sEsq[k0 + col];

        // B layout (dense f16 32x16): lane<16 -> col N=lane, elems = K d0+0..15;
        // lane>=16 -> col N=lane-16, elems = K d0+16..31. Contiguous 32B LDS reads.
        v16h b[8];
        #pragma unroll
        for (int c = 0; c < 8; ++c)
            b[c] = *(const v16h*)&curB[col * VQ_D + c * 32 + half * 16];

        // Two independent accumulation chains for matrix-pipe ILP
        v8f acc0, acc1;
        #pragma unroll
        for (int v = 0; v < 8; ++v) { acc0[v] = bias; acc1[v] = 0.0f; }
        #pragma unroll
        for (int c = 0; c < 4; ++c)
            acc0 = __builtin_amdgcn_wmma_f32_16x16x32_f16(
                false, a[c], false, b[c], (short)0, acc0, false, false);
        #pragma unroll
        for (int c = 4; c < 8; ++c)
            acc1 = __builtin_amdgcn_wmma_f32_16x16x32_f16(
                false, a[c], false, b[c], (short)0, acc1, false, false);

        const int code = k0 + col;
        #pragma unroll
        for (int v = 0; v < 8; ++v) {
            const float s = acc0[v] + acc1[v];
            if (s < minv[v]) { minv[v] = s; mini[v] = code; }
        }

        if (w == 0) __builtin_amdgcn_s_wait_tensorcnt(0);  // next tile resident
        __syncthreads();
    }

    // ---- Cross-lane argmin within each 16-lane half (xor tree, idx tie-break)
    #pragma unroll
    for (int v = 0; v < 8; ++v) {
        #pragma unroll
        for (int m = 8; m >= 1; m >>= 1) {
            float ov = __shfl_xor(minv[v], m, 32);
            int   oi = __shfl_xor(mini[v], m, 32);
            if (ov < minv[v] || (ov == minv[v] && oi < mini[v])) {
                minv[v] = ov; mini[v] = oi;
            }
        }
    }

    // lanes 0 and 16 publish final indices for rows 0..7 / 8..15 of the wave tile
    if (col == 0) {
        #pragma unroll
        for (int v = 0; v < 8; ++v) {
            const int m   = half * 8 + v;
            const int row = rowBase + m;
            const int bi  = mini[v];
            sRowIdx[w * 16 + m] = bi;
            out[ND + 3 + row] = (float)bi;     // encoding_indices (as f32)
            atomicAdd(&hist[bi], 1);
        }
    }
    __syncthreads();

    // ---- Gather quantized = emb[idx]*mask (f32), write quantized_st, loss partial
    float local = 0.0f;
    for (int m = 0; m < 16; ++m) {
        const int row = rowBase + m;
        const int idx = sRowIdx[w * 16 + m];
        const float mk = mask[row];
        const float* erow = emb + (size_t)idx * VQ_D;
        const float* xr   = x   + (size_t)row * VQ_D;
        float*       qr   = out + (size_t)row * VQ_D;
        const int o = lane * 8;
        #pragma unroll
        for (int j = 0; j < 8; ++j) {
            const float q  = erow[o + j] * mk;
            const float dx = q - xr[o + j];
            local += dx * dx;
            qr[o + j] = q;     // quantized_st forward value == quantized
        }
    }
    #pragma unroll
    for (int m = 16; m >= 1; m >>= 1) local += __shfl_xor(local, m, 32);
    if (lane == 0) sRed[w] = local;
    __syncthreads();
    if (tid == 0) {
        float s = 0.0f;
        #pragma unroll
        for (int i = 0; i < WAVES; ++i) s += sRed[i];
        partial[blockIdx.x] = s;    // deterministic per-WG slot (no float atomics)
    }
}

// ---------------------------------------------------------------------------
// Kernel 4: losses + perplexity
// ---------------------------------------------------------------------------
__global__ __launch_bounds__(256) void k_finalize(const int*   __restrict__ hist,
                                                  const float* __restrict__ partial,
                                                  float*       __restrict__ out) {
    __shared__ float sred[256];
    const int t = threadIdx.x;
    float ent = 0.0f;
    for (int k = t; k < VQ_K; k += 256) {
        const float p = (float)hist[k] * (1.0f / (float)VQ_N);
        ent -= p * logf(p + 1e-8f);
    }
    sred[t] = ent;
    __syncthreads();
    for (int s = 128; s > 0; s >>= 1) {
        if (t < s) sred[t] += sred[t + s];
        __syncthreads();
    }
    if (t == 0) {
        float ssum = 0.0f;
        for (int i = 0; i < VQ_N / ROWS_PER_WG; ++i) ssum += partial[i];
        const float mse = ssum / (float)((size_t)VQ_N * VQ_D);
        out[ND + 0] = 0.25f * mse;   // commitment_loss
        out[ND + 1] = mse;           // q_latent_loss (== e_latent numerically)
        out[ND + 2] = expf(sred[0]); // perplexity
    }
}

// ---------------------------------------------------------------------------
extern "C" void kernel_launch(void* const* d_in, const int* in_sizes, int n_in,
                              void* d_out, int out_size, void* d_ws, size_t ws_size,
                              hipStream_t stream) {
    const float* x    = (const float*)d_in[0];   // [N, D]
    const float* emb  = (const float*)d_in[1];   // [K, D]
    const float* mask = (const float*)d_in[2];   // [N]
    float* out = (float*)d_out;

    // workspace layout (256B-aligned slices)
    char* ws = (char*)d_ws;
    _Float16* emb_h   = (_Float16*)ws;                         // K*D*2 = 2 MB
    size_t off = (size_t)VQ_K * VQ_D * sizeof(_Float16);
    off = (off + 255) & ~(size_t)255;
    float* esq = (float*)(ws + off);                           // K*4
    off += (size_t)VQ_K * sizeof(float); off = (off + 255) & ~(size_t)255;
    int* hist = (int*)(ws + off);                              // K*4
    off += (size_t)VQ_K * sizeof(int);   off = (off + 255) & ~(size_t)255;
    float* partial = (float*)(ws + off);                       // (N/128)*4

    k_cvt_emb<<<(VQ_K * VQ_D) / 256, 256, 0, stream>>>(emb, emb_h);
    k_esq<<<VQ_K / 256, 256, 0, stream>>>(emb, esq, hist);
    vq_main<<<VQ_N / ROWS_PER_WG, 256, 0, stream>>>(x, emb, emb_h, esq, mask,
                                                    hist, partial, out);
    k_finalize<<<1, 256, 0, stream>>>(hist, partial, out);
}
